// GraphEdgeAttenNetworkLayers_edge_update_13305808683261
// MI455X (gfx1250) — matrix operand order repaired and checked by reading, not compiled, not run on hardware
//
#include <hip/hip_runtime.h>
#include <hip/hip_bf16.h>
#include <math.h>

// Problem constants (match reference)
#define NNODE 1024
#define EEDGE 4096
#define DN_   512
#define DE_   512
#define DA_   512
#define H_    8
#define DEP_  64
#define CCH   128   // DNP+DEP

typedef __attribute__((ext_vector_type(16))) __bf16 v16bf;
typedef __attribute__((ext_vector_type(8)))  __bf16 v8bf;
typedef __attribute__((ext_vector_type(8)))  float  v8f;

// ---------------------------------------------------------------- helpers
__device__ __forceinline__ __bf16 f2bf(float f) {
    unsigned u = __float_as_uint(f);
    u += 0x7FFFu + ((u >> 16) & 1u);      // round-to-nearest-even
    unsigned short hs = (unsigned short)(u >> 16);
    __bf16 r;
    __builtin_memcpy(&r, &hs, 2);
    return r;
}

__device__ __forceinline__ unsigned enc_f(float f) {
    unsigned u = __float_as_uint(f);
    return (u & 0x80000000u) ? ~u : (u | 0x80000000u);
}
__device__ __forceinline__ float dec_f(unsigned u) {
    unsigned b = (u & 0x80000000u) ? (u & 0x7FFFFFFFu) : ~u;
    return __uint_as_float(b);
}

// ---------------------------------------------------------------- WMMA GEMM
// C[M,N] = relu?( A[M,K](lda, bf16) @ Bt[N,K]^T(bf16) + bias[N] )
// One wave computes a 32x32 tile: 4 accumulators, A/B fragments reused.
// Dual epilogue: f32 out (Cf) and/or bf16 out (Ch), either may be null.
__global__ void __launch_bounds__(32)
gemm_bf16_wmma(const __bf16* __restrict__ A, int lda,
               const __bf16* __restrict__ Bt,
               const float* __restrict__ bias,
               float* __restrict__ Cf, __bf16* __restrict__ Ch,
               int ldc, int K, int relu)
{
    const int lane = threadIdx.x;        // 0..31, wave32
    const int hi   = lane >> 4;          // 0 or 1
    const int l16  = lane & 15;
    const int m0   = blockIdx.y * 32;
    const int n0   = blockIdx.x * 32;

    const __bf16* arow0 = A  + (size_t)(m0 + l16) * lda;
    const __bf16* arow1 = arow0 + (size_t)16 * lda;
    const __bf16* brow0 = Bt + (size_t)(n0 + l16) * K;
    const __bf16* brow1 = brow0 + (size_t)16 * K;

    v8f acc00 = {0.f,0.f,0.f,0.f,0.f,0.f,0.f,0.f};
    v8f acc01 = acc00, acc10 = acc00, acc11 = acc00;

    for (int k0 = 0; k0 < K; k0 += 32) {
        // A 16x32 fragment: lane<16 -> K[k0..k0+15]
        //                   lane>=16 -> K[k0+16..k0+31]
        const int ka = k0 + (hi ? 16 : 0);
        v8bf a0lo = *(const v8bf*)(arow0 + ka);
        v8bf a0hi = *(const v8bf*)(arow0 + ka + 8);
        v8bf a1lo = *(const v8bf*)(arow1 + ka);
        v8bf a1hi = *(const v8bf*)(arow1 + ka + 8);
        v16bf a0 = __builtin_shufflevector(a0lo, a0hi, 0,1,2,3,4,5,6,7,8,9,10,11,12,13,14,15);
        v16bf a1 = __builtin_shufflevector(a1lo, a1hi, 0,1,2,3,4,5,6,7,8,9,10,11,12,13,14,15);
        // B 32x16 fragment from Bt[N,K]: lane<16 -> K[k0..k0+15], lane>=16 -> K[k0+16..k0+31]
        const int kb = k0 + (hi ? 16 : 0);
        v16bf b0 = *(const v16bf*)(brow0 + kb);
        v16bf b1 = *(const v16bf*)(brow1 + kb);

        acc00 = __builtin_amdgcn_wmma_f32_16x16x32_bf16(false, a0, false, b0, (short)0, acc00, false, false);
        acc01 = __builtin_amdgcn_wmma_f32_16x16x32_bf16(false, a0, false, b1, (short)0, acc01, false, false);
        acc10 = __builtin_amdgcn_wmma_f32_16x16x32_bf16(false, a1, false, b0, (short)0, acc10, false, false);
        acc11 = __builtin_amdgcn_wmma_f32_16x16x32_bf16(false, a1, false, b1, (short)0, acc11, false, false);
    }

    // epilogue: lane l holds cols (n-tile)*16 + l16, rows (m-tile)*16 + hi*8 + r
    const int rb = hi ? 8 : 0;
#pragma unroll
    for (int t = 0; t < 4; ++t) {
        const v8f acc = (t == 0) ? acc00 : (t == 1) ? acc01 : (t == 2) ? acc10 : acc11;
        const int col = n0 + ((t & 1) ? 16 : 0) + l16;
        const int row0 = m0 + ((t & 2) ? 16 : 0) + rb;
        const float bv = bias ? bias[col] : 0.0f;
#pragma unroll
        for (int r = 0; r < 8; ++r) {
            float v = acc[r] + bv;
            if (relu) v = fmaxf(v, 0.0f);
            const size_t off = (size_t)(row0 + r) * ldc + col;
            if (Cf) Cf[off] = v;
            if (Ch) Ch[off] = f2bf(v);
        }
    }
}

// ---------------------------------------------------------------- transposes / converts
// f32 in[K,N] -> bf16 out[N,K]   (weights)
__global__ void convert_transpose(const float* __restrict__ in,
                                  __bf16* __restrict__ out, int K, int N)
{
    __shared__ float t[32][33];
    const int bx = blockIdx.x * 32;  // N
    const int by = blockIdx.y * 32;  // K
    const int x = bx + threadIdx.x;
    for (int dy = 0; dy < 32; dy += 8) {
        int y = by + threadIdx.y + dy;
        if (y < K && x < N) t[threadIdx.y + dy][threadIdx.x] = in[(size_t)y * N + x];
    }
    __syncthreads();
    const int ox = by + threadIdx.x;             // K index
    for (int dy = 0; dy < 32; dy += 8) {
        int oy = bx + threadIdx.y + dy;          // N index
        if (oy < N && ox < K) out[(size_t)oy * K + ox] = f2bf(t[threadIdx.x][threadIdx.y + dy]);
    }
}

// bf16 in[K,N] -> bf16 out[N,K]   (V matrices)
__global__ void transpose_bf16(const __bf16* __restrict__ in,
                               __bf16* __restrict__ out, int K, int N)
{
    __shared__ __bf16 t[32][33];
    const int bx = blockIdx.x * 32;  // N
    const int by = blockIdx.y * 32;  // K
    const int x = bx + threadIdx.x;
    for (int dy = 0; dy < 32; dy += 8) {
        int y = by + threadIdx.y + dy;
        if (y < K && x < N) t[threadIdx.y + dy][threadIdx.x] = in[(size_t)y * N + x];
    }
    __syncthreads();
    const int ox = by + threadIdx.x;
    for (int dy = 0; dy < 32; dy += 8) {
        int oy = bx + threadIdx.y + dy;
        if (oy < N && ox < K) out[(size_t)oy * K + ox] = t[threadIdx.x][threadIdx.y + dy];
    }
}

__global__ void f32_to_bf16(const float* __restrict__ in, __bf16* __restrict__ out, int n)
{
    int i = blockIdx.x * blockDim.x + threadIdx.x;
    if (i < n) out[i] = f2bf(in[i]);
}

// ---------------------------------------------------------------- softmax rows -> bf16 P
__global__ void __launch_bounds__(256)
softmax_rows_bf(float* __restrict__ S, __bf16* __restrict__ P, int Nn, float scale)
{
    __shared__ float red[256];
    float* p = S + (size_t)blockIdx.x * Nn;
    __bf16* q = P + (size_t)blockIdx.x * Nn;
    float m = -INFINITY;
    for (int j = threadIdx.x; j < Nn; j += 256) m = fmaxf(m, p[j] * scale);
    red[threadIdx.x] = m; __syncthreads();
    for (int s = 128; s > 0; s >>= 1) {
        if (threadIdx.x < s) red[threadIdx.x] = fmaxf(red[threadIdx.x], red[threadIdx.x + s]);
        __syncthreads();
    }
    m = red[0]; __syncthreads();
    float sum = 0.f;
    for (int j = threadIdx.x; j < Nn; j += 256) {
        float e = __expf(p[j] * scale - m);
        p[j] = e; sum += e;
    }
    red[threadIdx.x] = sum; __syncthreads();
    for (int s = 128; s > 0; s >>= 1) {
        if (threadIdx.x < s) red[threadIdx.x] += red[threadIdx.x + s];
        __syncthreads();
    }
    const float inv = 1.0f / red[0];
    for (int j = threadIdx.x; j < Nn; j += 256) q[j] = f2bf(p[j] * inv);
}

// ---------------------------------------------------------------- gathers / concats
__global__ void gather_concat_bf(const float* __restrict__ x, const int* __restrict__ ei,
                                 __bf16* __restrict__ out)
{
    int idx = blockIdx.x * blockDim.x + threadIdx.x;
    const int W = 2 * DN_;
    if (idx >= EEDGE * W) return;
    int e = idx / W, d = idx % W;
    int node = (d < DN_) ? ei[e] : ei[EEDGE + e];
    int dd   = (d < DN_) ? d : d - DN_;
    out[idx] = f2bf(x[(size_t)node * DN_ + dd]);
}

__global__ void concat3_bf(const __bf16* __restrict__ a, const __bf16* __restrict__ b,
                           const __bf16* __restrict__ c, __bf16* __restrict__ out)
{
    int idx = blockIdx.x * blockDim.x + threadIdx.x;
    const int W = 3 * DE_;
    if (idx >= EEDGE * W) return;
    int e = idx / W, d = idx % W;
    __bf16 v;
    if (d < DE_)            v = a[(size_t)e * DE_ + d];
    else if (d < 2 * DE_)   v = b[(size_t)e * DE_ + d - DE_];
    else                    v = c[(size_t)e * DE_ + d - 2 * DE_];
    out[idx] = v;
}

__global__ void concat2_bf(const float* __restrict__ a, const float* __restrict__ b,
                           __bf16* __restrict__ out)
{
    int idx = blockIdx.x * blockDim.x + threadIdx.x;
    const int W = DN_ + DA_;
    if (idx >= NNODE * W) return;
    int r = idx / W, d = idx % W;
    float v = (d < DN_) ? a[(size_t)r * DN_ + d] : b[(size_t)r * DA_ + d - DN_];
    out[idx] = f2bf(v);
}

// ---------------------------------------------------------------- fused per-edge conv attention
__global__ void __launch_bounds__(128)
nn_att_kernel(const float* __restrict__ Qp, const float* __restrict__ Kp,
              const float* __restrict__ Vj,
              const float* __restrict__ Wa1, const float* __restrict__ ba1,
              const float* __restrict__ ga1, const float* __restrict__ beta1,
              const float* __restrict__ Wa2, const float* __restrict__ ba2,
              float* __restrict__ value, float* __restrict__ prob)
{
    const int e = blockIdx.x;
    const int t = threadIdx.x;
    __shared__ float a[CCH], h1[CCH], att[DEP_], red[2];
    const float bnscale = rsqrtf(1.0f + 1e-5f);

    for (int h = 0; h < H_; ++h) {
        a[t] = (t < 64) ? Qp[(size_t)e * DE_ + t * H_ + h]
                        : Kp[(size_t)e * DE_ + (t - 64) * H_ + h];
        __syncthreads();
        {
            float s = ba1[t];
            const float* wr = Wa1 + (size_t)t * CCH;
            for (int c = 0; c < CCH; ++c) s += a[c] * wr[c];
            s = s * (ga1[t] * bnscale) + beta1[t];
            h1[t] = fmaxf(s, 0.0f);
        }
        __syncthreads();
        if (t < DEP_) {
            float s = ba2[t];
            const float* wr = Wa2 + (size_t)t * CCH;
            for (int o = 0; o < CCH; ++o) s += h1[o] * wr[o];
            att[t] = s * 0.125f;   // 1/sqrt(DEP)
        }
        __syncthreads();
        if (t == 0) {
            float m = -INFINITY;
            for (int o = 0; o < DEP_; ++o) m = fmaxf(m, att[o]);
            red[0] = m;
        }
        __syncthreads();
        if (t < DEP_) att[t] = __expf(att[t] - red[0]);
        __syncthreads();
        if (t == 0) {
            float su = 0.f;
            for (int o = 0; o < DEP_; ++o) su += att[o];
            red[1] = 1.0f / su;
        }
        __syncthreads();
        if (t < DEP_) {
            float pn = att[t] * red[1];
            prob[((size_t)e * DEP_ + t) * H_ + h] = pn;
            int d = t * H_ + h;
            value[(size_t)e * DA_ + d] = pn * Vj[(size_t)e * DA_ + d];
        }
        __syncthreads();
    }
}

// ---------------------------------------------------------------- scatter max
__global__ void fill_u32(unsigned* __restrict__ p, unsigned v, int n)
{
    int i = blockIdx.x * blockDim.x + threadIdx.x;
    if (i < n) p[i] = v;
}

__global__ void scatter_max_enc(const float* __restrict__ value, const int* __restrict__ ei,
                                unsigned* __restrict__ enc)
{
    int idx = blockIdx.x * blockDim.x + threadIdx.x;
    if (idx >= EEDGE * DA_) return;
    int e = idx / DA_, d = idx % DA_;
    atomicMax(&enc[(size_t)ei[e] * DA_ + d], enc_f(value[idx]));
}

__global__ void decode_agg(const unsigned* __restrict__ enc, float* __restrict__ agg, int n)
{
    int i = blockIdx.x * blockDim.x + threadIdx.x;
    if (i >= n) return;
    unsigned u = enc[i];
    agg[i] = (u == 0u) ? 0.0f : dec_f(u);   // empty segment -> 0 (PyG convention)
}

// ---------------------------------------------------------------- host side
static inline void gemm(hipStream_t s, const __bf16* A, int lda, const __bf16* Bt,
                        const float* bias, float* Cf, __bf16* Ch, int ldc,
                        int M, int N, int K, bool relu)
{
    dim3 g(N / 32, M / 32), b(32);
    gemm_bf16_wmma<<<g, b, 0, s>>>(A, lda, Bt, bias, Cf, Ch, ldc, K, relu ? 1 : 0);
}

static inline void wT(hipStream_t s, const float* in, __bf16* out, int K, int N)
{
    dim3 g((N + 31) / 32, (K + 31) / 32), b(32, 8);
    convert_transpose<<<g, b, 0, s>>>(in, out, K, N);
}

static inline void tBF(hipStream_t s, const __bf16* in, __bf16* out, int K, int N)
{
    dim3 g((N + 31) / 32, (K + 31) / 32), b(32, 8);
    transpose_bf16<<<g, b, 0, s>>>(in, out, K, N);
}

static inline int cdiv(int a, int b) { return (a + b - 1) / b; }

// sorted-key order of each layer's param dict (JAX pytree convention)
enum {
    iWa1 = 0, iWa2, iWc, iWe1, iWe2, iWk1, iWk2, iWpk, iWpq, iWpv,
    iWq1, iWq2, iWu1, iWu2, iWv1, iWv2,
    iba1, iba2, ibc, ibe1, ibe2, ibeta1, ibk1, ibk2, ibpk, ibpq, ibpv,
    ibq1, ibq2, ibu1, ibu2, ibv1, ibv2, iga1, NPARAM
};

extern "C" void kernel_launch(void* const* d_in, const int* in_sizes, int n_in,
                              void* d_out, int out_size, void* d_ws, size_t ws_size,
                              hipStream_t stream)
{
    (void)in_sizes; (void)n_in; (void)out_size; (void)ws_size;
    const float* x_in  = (const float*)d_in[0];
    const float* ef_in = (const float*)d_in[1];
    const int*   ei    = (const int*)d_in[2];
    auto P = [&](int li, int idx) -> const float* {
        return (const float*)d_in[3 + li * NPARAM + idx];
    };

    // bump allocator on workspace
    char* wsp = (char*)d_ws;
    auto allocB = [&](size_t bytes) -> void* {
        void* p = (void*)wsp;
        wsp += (bytes + 255) & ~(size_t)255;
        return p;
    };
    auto aF = [&](size_t n) -> float*  { return (float*)allocB(n * 4); };
    auto aH = [&](size_t n) -> __bf16* { return (__bf16*)allocB(n * 2); };

    // bf16 buffers
    __bf16* xi_xj  = aH((size_t)EEDGE * 2 * DN_);
    __bf16* comb   = aH((size_t)EEDGE * DE_);
    __bf16* qbf    = aH((size_t)EEDGE * DE_);
    __bf16* kbf    = aH((size_t)EEDGE * DE_);
    __bf16* vbf    = aH((size_t)EEDGE * DE_);
    __bf16* vtb    = aH((size_t)DE_ * EEDGE);
    __bf16* Pbf    = aH((size_t)EEDGE * EEDGE);
    __bf16* o1bf   = aH((size_t)EEDGE * DE_);
    __bf16* o2bf   = aH((size_t)EEDGE * DE_);
    __bf16* efbf0  = aH((size_t)EEDGE * DE_);
    __bf16* efbf1  = aH((size_t)EEDGE * DE_);
    __bf16* cat3bf = aH((size_t)EEDGE * 3 * DE_);
    __bf16* hbf    = aH((size_t)EEDGE * 2 * DE_);
    __bf16* xcatbf = aH((size_t)NNODE * (DN_ + DA_));
    __bf16* ubf    = aH((size_t)NNODE * (DN_ + DA_));
    __bf16* wtb    = aH((size_t)1536 * 1024);     // largest transposed weight (We1)
    // f32 buffers
    float* S     = aF((size_t)EEDGE * EEDGE);     // scores; qpf/kpf/vjf alias it afterwards
    float* efbuf = aF((size_t)EEDGE * DE_);
    float* xbuf  = aF((size_t)NNODE * DN_);
    float* val   = aF((size_t)EEDGE * DA_);
    unsigned* enc = (unsigned*)allocB((size_t)NNODE * DA_ * 4);
    float* agg   = aF((size_t)NNODE * DA_);
    // node-att projections carve f32 space from S (scores dead by then)
    float* qpf = S;
    float* kpf = S + (size_t)EEDGE * DE_;
    float* vjf = S + (size_t)2 * EEDGE * DE_;

    float* out_x  = (float*)d_out;                                      // [1024,512]
    float* out_ef = (float*)d_out + (size_t)NNODE * DN_;                // [4096,512]
    float* out_pr = out_ef + (size_t)EEDGE * DE_;                       // [2,4096,64,8]

    const float sm_scale = 1.0f / sqrtf((float)DE_);

    // ef (layer-0 input) -> bf16 once
    f32_to_bf16<<<cdiv(EEDGE * DE_, 256), 256, 0, stream>>>(ef_in, efbf0, EEDGE * DE_);

    for (int li = 0; li < 2; ++li) {
        const float*  xc   = li ? xbuf  : x_in;
        const float*  efcf = li ? efbuf : ef_in;   // f32 view of current ef (unused by GEMMs)
        const __bf16* efbf = li ? efbf1 : efbf0;   // bf16 view of current ef
        (void)efcf;
        float* xout  = (li == 1) ? out_x  : xbuf;
        float* efout = (li == 1) ? out_ef : efbuf;
        float* prout = out_pr + (size_t)li * EEDGE * DEP_ * H_;
        const bool relu_out = (li < 1);

        // x_i || x_j  (f32 gather -> bf16)
        gather_concat_bf<<<cdiv(EEDGE * 2 * DN_, 256), 256, 0, stream>>>(xc, ei, xi_xj);

        // combined = (x_i||x_j) @ Wc + bc        (bf16 out only; feeds GEMMs)
        wT(stream, P(li, iWc), wtb, 2 * DN_, DE_);
        gemm(stream, xi_xj, 2 * DN_, wtb, P(li, ibc), nullptr, comb, DE_, EEDGE, DE_, 2 * DN_, false);

        // ---- cross attention 1: combined queries ef
        wT(stream, P(li, iWq1), wtb, DE_, DE_);
        gemm(stream, comb, DE_, wtb, P(li, ibq1), nullptr, qbf, DE_, EEDGE, DE_, DE_, false);
        wT(stream, P(li, iWk1), wtb, DE_, DE_);
        gemm(stream, efbf, DE_, wtb, P(li, ibk1), nullptr, kbf, DE_, EEDGE, DE_, DE_, false);
        wT(stream, P(li, iWv1), wtb, DE_, DE_);
        gemm(stream, efbf, DE_, wtb, P(li, ibv1), nullptr, vbf, DE_, EEDGE, DE_, DE_, false);
        gemm(stream, qbf, DE_, kbf, nullptr, S, nullptr, EEDGE, EEDGE, EEDGE, DE_, false);
        softmax_rows_bf<<<EEDGE, 256, 0, stream>>>(S, Pbf, EEDGE, sm_scale);
        tBF(stream, vbf, vtb, EEDGE, DE_);
        gemm(stream, Pbf, EEDGE, vtb, nullptr, nullptr, o1bf, DE_, EEDGE, DE_, EEDGE, false);

        // ---- cross attention 2: ef queries combined
        wT(stream, P(li, iWq2), wtb, DE_, DE_);
        gemm(stream, efbf, DE_, wtb, P(li, ibq2), nullptr, qbf, DE_, EEDGE, DE_, DE_, false);
        wT(stream, P(li, iWk2), wtb, DE_, DE_);
        gemm(stream, comb, DE_, wtb, P(li, ibk2), nullptr, kbf, DE_, EEDGE, DE_, DE_, false);
        wT(stream, P(li, iWv2), wtb, DE_, DE_);
        gemm(stream, comb, DE_, wtb, P(li, ibv2), nullptr, vbf, DE_, EEDGE, DE_, DE_, false);
        gemm(stream, qbf, DE_, kbf, nullptr, S, nullptr, EEDGE, EEDGE, EEDGE, DE_, false);
        softmax_rows_bf<<<EEDGE, 256, 0, stream>>>(S, Pbf, EEDGE, sm_scale);
        tBF(stream, vbf, vtb, EEDGE, DE_);
        gemm(stream, Pbf, EEDGE, vtb, nullptr, nullptr, o2bf, DE_, EEDGE, DE_, EEDGE, false);

        // ---- edge update MLP
        concat3_bf<<<cdiv(EEDGE * 3 * DE_, 256), 256, 0, stream>>>(efbf, o1bf, o2bf, cat3bf);
        wT(stream, P(li, iWe1), wtb, 3 * DE_, 2 * DE_);
        gemm(stream, cat3bf, 3 * DE_, wtb, P(li, ibe1), nullptr, hbf, 2 * DE_, EEDGE, 2 * DE_, 3 * DE_, true);
        wT(stream, P(li, iWe2), wtb, 2 * DE_, DE_);
        gemm(stream, hbf, 2 * DE_, wtb, P(li, ibe2), efout, (li == 0) ? efbf1 : nullptr,
             DE_, EEDGE, DE_, 2 * DE_, relu_out);

        // ---- per-edge multi-head attention for node aggregation (f32 outputs)
        wT(stream, P(li, iWpq), wtb, DN_, DN_);
        gemm(stream, xi_xj, 2 * DN_, wtb, P(li, ibpq), qpf, nullptr, DN_, EEDGE, DN_, DN_, false);
        wT(stream, P(li, iWpk), wtb, DE_, DE_);
        gemm(stream, efbf, DE_, wtb, P(li, ibpk), kpf, nullptr, DE_, EEDGE, DE_, DE_, false);
        wT(stream, P(li, iWpv), wtb, DN_, DA_);
        gemm(stream, xi_xj + DN_, 2 * DN_, wtb, P(li, ibpv), vjf, nullptr, DA_, EEDGE, DA_, DN_, false);
        nn_att_kernel<<<EEDGE, 128, 0, stream>>>(qpf, kpf, vjf,
                                                 P(li, iWa1), P(li, iba1), P(li, iga1), P(li, ibeta1),
                                                 P(li, iWa2), P(li, iba2), val, prout);

        // ---- scatter-max aggregation at index i
        fill_u32<<<cdiv(NNODE * DA_, 256), 256, 0, stream>>>(enc, 0u, NNODE * DA_);
        scatter_max_enc<<<cdiv(EEDGE * DA_, 256), 256, 0, stream>>>(val, ei, enc);
        decode_agg<<<cdiv(NNODE * DA_, 256), 256, 0, stream>>>(enc, agg, NNODE * DA_);

        // ---- node update MLP
        concat2_bf<<<cdiv(NNODE * (DN_ + DA_), 256), 256, 0, stream>>>(xc, agg, xcatbf);
        wT(stream, P(li, iWu1), wtb, DN_ + DA_, DN_ + DA_);
        gemm(stream, xcatbf, DN_ + DA_, wtb, P(li, ibu1), nullptr, ubf, DN_ + DA_,
             NNODE, DN_ + DA_, DN_ + DA_, true);
        wT(stream, P(li, iWu2), wtb, DN_ + DA_, DN_);
        gemm(stream, ubf, DN_ + DA_, wtb, P(li, ibu2), xout, nullptr, DN_,
             NNODE, DN_, DN_ + DA_, relu_out);
    }
}
